// SwinTransformerLayer_78434692759796
// MI455X (gfx1250) — matrix operand order repaired and checked by reading, not compiled, not run on hardware
//
#include <hip/hip_runtime.h>
#include <hip/hip_bf16.h>

typedef _Float16 h16;
typedef __attribute__((ext_vector_type(16))) _Float16 v16h;
typedef __attribute__((ext_vector_type(8)))  _Float16 v8h;
typedef __attribute__((ext_vector_type(8)))  float    v8f;

#define B_      8
#define C_      192
#define H_      128
#define W_      128
#define HW_     (H_*W_)
#define WSZ     8
#define SHIFT_  4
#define NTOK    64
#define NWIN    256
#define HEADS_  6
#define HD_     32
#define EMB_    256
#define HID_    768
#define QKV_SCALE 0.17677669529663687f   // 1/sqrt(32)

// Padded LDS leading dimensions (bank-conflict avoidance, rows stay 16B-aligned)
#define XLD 200   // 64x192 tile rows (h16): 100 words/row, 36*r mod 64 distinct
#define QLD 392   // 64x384 qk rows  (h16): 196 words/row, 4*r mod 64 distinct
#define VLD 72    // vT / probs rows (h16): 36 words/row, distinct banks
#define PLD 72
#define SLD 66    // S rows (f32): banks (2r+m) mod 64 -> conflict-free softmax
#define ALD 776   // 64x768 act rows (h16): 388 words/row, 4*r mod 64 distinct

// ---------------- WMMA fragment helpers (gfx1250 layouts, ISA 7.12.2) ----------

// A operand: M x K (K=32, f16). lane l: row m=l&15; holds K = kb..kb+7 and
// kb+16..kb+23 with kb = 8*(l>>4). Source is row-major (ld = row stride in h16).
__device__ inline v16h load_fragA(const h16* base, int ld, int lane) {
  int r  = lane & 15;
  int kb = (lane >> 4) << 3;
  const h16* row = base + r * ld + kb;
  v8h lo = *(const v8h*)(row);
  v8h hi = *(const v8h*)(row + 16);
  v16h f;
#pragma unroll
  for (int i = 0; i < 8; ++i) { f[i] = lo[i]; f[i + 8] = hi[i]; }
  return f;
}

// B operand: K x N (K=32, f16), stored as (N x K) row-major ("transposed B",
// i.e. weights in (out_features, in_features)). lane l: col n=l&15; holds
// K = kb..kb+7 and kb+16..kb+23 with kb = 8*(l>>4).
__device__ inline v16h load_fragB(const h16* base, int ld, int lane) {
  int n  = lane & 15;
  int kb = (lane >> 4) << 3;
  const h16* row = base + n * ld + kb;
  v8h lo = *(const v8h*)(row);
  v8h hi = *(const v8h*)(row + 16);
  v16h f;
#pragma unroll
  for (int i = 0; i < 8; ++i) { f[i] = lo[i]; f[i + 8] = hi[i]; }
  return f;
}

__device__ inline v8f wmma_f16(v16h a, v16h b, v8f c) {
  return __builtin_amdgcn_wmma_f32_16x16x32_f16(false, a, false, b, (short)0, c,
                                                false, false);
}

// ---------------- Kernel 0: convert weights to f16 (out,in layout kept) -------
__global__ void k_convert(const float* qkv_w, const float* proj_w,
                          const float* w1, const float* w2,
                          h16* qkv_h, h16* proj_h, h16* w1h, h16* w2h) {
  int i = blockIdx.x * blockDim.x + threadIdx.x;
  const int n0 = 576 * 192, n1 = 192 * 192, n2 = 768 * 192, n3 = 192 * 768;
  if (i < n0) qkv_h[i] = (h16)qkv_w[i];
  else if (i < n0 + n1) proj_h[i - n0] = (h16)proj_w[i - n0];
  else if (i < n0 + n1 + n2) w1h[i - n0 - n1] = (h16)w1[i - n0 - n1];
  else if (i < n0 + n1 + n2 + n3) w2h[i - n0 - n1 - n2] = (h16)w2[i - n0 - n1 - n2];
}

// ---------------- Kernel 1: FiLM  gb[b,j] = cond[b,:] . film_w[j,:] + film_b --
__global__ void k_film(const float* cond, const float* fw, const float* fb,
                       float* gb) {
  int i = blockIdx.x * blockDim.x + threadIdx.x;
  if (i >= B_ * 2 * C_) return;
  int b = i / (2 * C_), j = i % (2 * C_);
  float s = fb[j];
  for (int k = 0; k < EMB_; ++k) s += cond[b * EMB_ + k] * fw[j * EMB_ + k];
  gb[i] = s;
}

// ---------------- Kernel 2/4: per-(b,c) instance-norm stats -------------------
__global__ void k_stats(const float* x, float* mean, float* rstd) {
  __shared__ float red[256], red2[256];
  int bc = blockIdx.x;
  const float* p = x + (size_t)bc * HW_;
  float s = 0.f, s2 = 0.f;
  for (int i = threadIdx.x; i < HW_; i += 256) {
    float v = p[i]; s += v; s2 += v * v;
  }
  red[threadIdx.x] = s; red2[threadIdx.x] = s2;
  __syncthreads();
  for (int o = 128; o > 0; o >>= 1) {
    if ((int)threadIdx.x < o) {
      red[threadIdx.x]  += red[threadIdx.x + o];
      red2[threadIdx.x] += red2[threadIdx.x + o];
    }
    __syncthreads();
  }
  if (threadIdx.x == 0) {
    float m = red[0] * (1.f / HW_);
    float v = red2[0] * (1.f / HW_) - m * m;
    mean[bc] = m;
    rstd[bc] = rsqrtf(v + 1e-5f);
  }
}

// ---------------- Kernel 3: fused shifted-window attention --------------------
// One block per (b, window). 256 threads = 8 waves.
// LDS (padded):
//   xtile 64xXLD h16 @0       (25600)   [reused as attn_out]
//   qk    64xQLD h16 @25600   (50176)
//   vT   192xVLD h16 @75776   (27648)
//   S   6x64xSLD f32 @103424  (101376)
//   probs 6x64xPLD h16 @204800 (55296)   => 260096 B total
__global__ void k_attn(const float* __restrict__ x, const float* mean,
                       const float* rstd, const h16* __restrict__ qkv_wh,
                       const float* qkv_b, const h16* __restrict__ proj_wh,
                       const float* proj_b, const float* __restrict__ rel_bias,
                       float* __restrict__ x2) {
  extern __shared__ char smem[];
  h16*   xtile = (h16*)smem;
  h16*   qk    = (h16*)(smem + 25600);
  h16*   vT    = (h16*)(smem + 75776);
  float* S     = (float*)(smem + 103424);
  h16*   probs = (h16*)(smem + 204800);

  int blk = blockIdx.x;
  int b = blk >> 8, wi = blk & 255;
  int wr = wi >> 4, wc = wi & 15;
  int tid = threadIdx.x, lane = tid & 31, wave = tid >> 5;

  // warm caches for the weight tables (global_prefetch_b8 path)
  for (int i = tid; i < 1728; i += 256) __builtin_prefetch(qkv_wh + i * 64, 0, 2);
  for (int i = tid; i < 576;  i += 256) __builtin_prefetch(proj_wh + i * 64, 0, 2);

  // (a) gather + instance-normalize the (shift-rolled) window tile
  for (int i = tid; i < NTOK * C_; i += 256) {
    int n = i / C_, c = i % C_;
    int hs = ((wr << 3) + (n >> 3) + SHIFT_) & 127;
    int ws = ((wc << 3) + (n & 7) + SHIFT_) & 127;
    size_t bc = (size_t)b * C_ + c;
    float v = (x[bc * HW_ + hs * W_ + ws] - mean[bc]) * rstd[bc];
    xtile[n * XLD + c] = (h16)v;
  }
  __syncthreads();

  // (b) QKV GEMM: 64x576 in three branch-free passes (Q / K / V tiles).
  // --- Q tiles (output cols 0..191), pre-scaled by 1/sqrt(hd) ---
  for (int t = wave; t < 48; t += 8) {
    int mt = t & 3, nt = t >> 2;
    v8f acc = {};
#pragma unroll
    for (int kt = 0; kt < 6; ++kt) {
      v16h a  = load_fragA(xtile + mt * 16 * XLD + kt * 32, XLD, lane);
      v16h bm = load_fragB(qkv_wh + (size_t)nt * 16 * C_ + kt * 32, C_, lane);
      acc = wmma_f16(a, bm, acc);
    }
    int nn = lane & 15, hi = lane >> 4;
    int f = nt * 16 + nn;
    float bias = qkv_b[f];
#pragma unroll
    for (int i = 0; i < 8; ++i) {
      int m = mt * 16 + i + (hi << 3);
      qk[m * QLD + f] = (h16)((acc[i] + bias) * QKV_SCALE);
    }
  }
  // --- K tiles (output cols 192..383) ---
  for (int t = wave; t < 48; t += 8) {
    int mt = t & 3, nt = t >> 2;
    v8f acc = {};
#pragma unroll
    for (int kt = 0; kt < 6; ++kt) {
      v16h a  = load_fragA(xtile + mt * 16 * XLD + kt * 32, XLD, lane);
      v16h bm = load_fragB(qkv_wh + (size_t)(nt + 12) * 16 * C_ + kt * 32, C_, lane);
      acc = wmma_f16(a, bm, acc);
    }
    int nn = lane & 15, hi = lane >> 4;
    int f = 192 + nt * 16 + nn;
    float bias = qkv_b[f];
#pragma unroll
    for (int i = 0; i < 8; ++i) {
      int m = mt * 16 + i + (hi << 3);
      qk[m * QLD + f] = (h16)(acc[i] + bias);
    }
  }
  // --- V tiles (output cols 384..575), stored transposed (d, token) ---
  for (int t = wave; t < 48; t += 8) {
    int mt = t & 3, nt = t >> 2;
    v8f acc = {};
#pragma unroll
    for (int kt = 0; kt < 6; ++kt) {
      v16h a  = load_fragA(xtile + mt * 16 * XLD + kt * 32, XLD, lane);
      v16h bm = load_fragB(qkv_wh + (size_t)(nt + 24) * 16 * C_ + kt * 32, C_, lane);
      acc = wmma_f16(a, bm, acc);
    }
    int nn = lane & 15, hi = lane >> 4;
    int ff = nt * 16 + nn;               // 0..191 within V block
    float bias = qkv_b[384 + ff];
    int head = ff >> 5, d = ff & 31;
#pragma unroll
    for (int i = 0; i < 8; ++i) {
      int m = mt * 16 + i + (hi << 3);
      vT[(head * 32 + d) * VLD + m] = (h16)(acc[i] + bias);
    }
  }
  __syncthreads();

  // (c) S = Q K^T: 6 heads x 16 tiles = 96 tiles over all 8 waves (K=32=hd)
  for (int t = wave; t < 96; t += 8) {
    int head = t >> 4, r = t & 15, mt = r & 3, nt = r >> 2;
    const h16* qb = qk + head * 32;
    const h16* kb = qk + 192 + head * 32;
    float* Sh = S + head * 64 * SLD;
    v8f acc = {};
    v16h a  = load_fragA(qb + mt * 16 * QLD, QLD, lane);
    v16h bm = load_fragB(kb + nt * 16 * QLD, QLD, lane);
    acc = wmma_f16(a, bm, acc);
    int nn = lane & 15, hi = lane >> 4;
#pragma unroll
    for (int i = 0; i < 8; ++i)
      Sh[(mt * 16 + i + (hi << 3)) * SLD + nt * 16 + nn] = acc[i];
  }
  __syncthreads();

  // (d) relative-position bias + shift mask + softmax (row per thread)
  for (int r = tid; r < HEADS_ * 64; r += 256) {
    int head = r >> 6, n = r & 63;
    float* Sr = S + (head * 64 + n) * SLD;
    h16*   pr = probs + (head * 64 + n) * PLD;
    int hn = (wr << 3) + (n >> 3), wn = (wc << 3) + (n & 7);
    int img_n = ((hn >= H_ - WSZ) + (hn >= H_ - SHIFT_)) * 3 +
                ((wn >= W_ - WSZ) + (wn >= W_ - SHIFT_));
    int rn0 = n >> 3, rn1 = n & 7;
    float mx = -1e30f;
    for (int m = 0; m < 64; ++m) {
      int hm = (wr << 3) + (m >> 3), wm = (wc << 3) + (m & 7);
      int img_m = ((hm >= H_ - WSZ) + (hm >= H_ - SHIFT_)) * 3 +
                  ((wm >= W_ - WSZ) + (wm >= W_ - SHIFT_));
      int ri = (rn0 - (m >> 3) + 7) * 15 + (rn1 - (m & 7) + 7);
      float v = Sr[m] + rel_bias[ri * HEADS_ + head] +
                ((img_n != img_m) ? -1e9f : 0.f);
      Sr[m] = v;
      mx = fmaxf(mx, v);
    }
    float sum = 0.f;
    for (int m = 0; m < 64; ++m) { float e = __expf(Sr[m] - mx); sum += e; Sr[m] = e; }
    float inv = 1.f / sum;
    for (int m = 0; m < 64; ++m) pr[m] = (h16)(Sr[m] * inv);
  }
  __syncthreads();

  // (e) O = probs @ V: 6 heads x (4x2) tiles = 48 tiles over 8 waves, K=64.
  //     Output tile -> xtile (reused as attn_out, 64 x C_ with XLD stride).
  for (int t = wave; t < 48; t += 8) {
    int head = t >> 3, r = t & 7, mt = r & 3, nt = r >> 2;
    const h16* pb = probs + head * 64 * PLD;
    const h16* vb = vT + head * 32 * VLD;
    v8f acc = {};
#pragma unroll
    for (int kt = 0; kt < 2; ++kt) {
      v16h a  = load_fragA(pb + mt * 16 * PLD + kt * 32, PLD, lane);
      v16h bm = load_fragB(vb + nt * 16 * VLD + kt * 32, VLD, lane);
      acc = wmma_f16(a, bm, acc);
    }
    int nn = lane & 15, hi = lane >> 4;
#pragma unroll
    for (int i = 0; i < 8; ++i)
      xtile[(mt * 16 + i + (hi << 3)) * XLD + head * 32 + nt * 16 + nn] =
          (h16)acc[i];
  }
  __syncthreads();

  // (f) projection GEMM + un-shift scatter + residual: x2 = x + proj(o)
  for (int t = wave; t < 48; t += 8) {
    int mt = t & 3, nt = t >> 2;
    v8f acc = {};
#pragma unroll
    for (int kt = 0; kt < 6; ++kt) {
      v16h a  = load_fragA(xtile + mt * 16 * XLD + kt * 32, XLD, lane);
      v16h bm = load_fragB(proj_wh + (size_t)nt * 16 * C_ + kt * 32, C_, lane);
      acc = wmma_f16(a, bm, acc);
    }
    int nn = lane & 15, hi = lane >> 4;
    int f = nt * 16 + nn;
    float pb2 = proj_b[f];
#pragma unroll
    for (int i = 0; i < 8; ++i) {
      int m = mt * 16 + i + (hi << 3);
      int hh = ((wr << 3) + (m >> 3) + SHIFT_) & 127;
      int ww = ((wc << 3) + (m & 7) + SHIFT_) & 127;
      size_t idx = ((size_t)b * C_ + f) * HW_ + hh * W_ + ww;
      x2[idx] = x[idx] + acc[i] + pb2;
    }
  }
}

// ---------------- Kernel 5: fused FiLM-norm + MLP ----------------------------
// One block per 64 pixels. LDS: hm 64xXLD h16 @0 (25600) | act 64xALD h16
// @25600 (99328)  => 124928 B
__global__ void k_mlp(const float* __restrict__ x2, const float* mean2,
                      const float* rstd2, const float* gb,
                      const h16* __restrict__ w1h, const float* b1,
                      const h16* __restrict__ w2h, const float* b2,
                      float* __restrict__ out) {
  extern __shared__ char smem[];
  h16* hm  = (h16*)smem;
  h16* act = (h16*)(smem + 25600);
  int tid = threadIdx.x, lane = tid & 31, wave = tid >> 5;
  size_t p0 = (size_t)blockIdx.x * 64;

  // warm caches for the weight tables
  for (int i = tid; i < 2304; i += 256) __builtin_prefetch(w1h + i * 64, 0, 2);
  for (int i = tid; i < 2304; i += 256) __builtin_prefetch(w2h + i * 64, 0, 2);

  // (a) modulated normalized tile
  for (int i = tid; i < 64 * C_; i += 256) {
    int pl = i / C_, c = i % C_;
    size_t p = p0 + pl;
    int b  = (int)(p / HW_);
    int pp = (int)(p % HW_);
    size_t bc = (size_t)b * C_ + c;
    float v = (x2[bc * HW_ + pp] - mean2[bc]) * rstd2[bc];
    v = v * (1.f + gb[b * 2 * C_ + c]) + gb[b * 2 * C_ + C_ + c];
    hm[pl * XLD + c] = (h16)v;
  }
  __syncthreads();

  // (b) GEMM1 (192->768) + tanh-gelu: 4 x 48 tiles, K=192
  for (int t = wave; t < 192; t += 8) {
    int mt = t & 3, nt = t >> 2;
    v8f acc = {};
#pragma unroll
    for (int kt = 0; kt < 6; ++kt) {
      v16h a  = load_fragA(hm + mt * 16 * XLD + kt * 32, XLD, lane);
      v16h bm = load_fragB(w1h + (size_t)nt * 16 * C_ + kt * 32, C_, lane);
      acc = wmma_f16(a, bm, acc);
    }
    int nn = lane & 15, hi = lane >> 4;
    int f = nt * 16 + nn;
    float bb = b1[f];
#pragma unroll
    for (int i = 0; i < 8; ++i) {
      float v = acc[i] + bb;
      float g = 0.5f * v * (1.f + tanhf(0.7978845608f * (v + 0.044715f * v * v * v)));
      act[(mt * 16 + i + (hi << 3)) * ALD + f] = (h16)g;
    }
  }
  __syncthreads();

  // (c) GEMM2 (768->192) + residual: 4 x 12 tiles, K=768 (24 wmma steps)
  for (int t = wave; t < 48; t += 8) {
    int mt = t & 3, nt = t >> 2;
    v8f acc = {};
#pragma unroll
    for (int kt = 0; kt < 24; ++kt) {
      v16h a  = load_fragA(act + mt * 16 * ALD + kt * 32, ALD, lane);
      v16h bm = load_fragB(w2h + (size_t)nt * 16 * HID_ + kt * 32, HID_, lane);
      acc = wmma_f16(a, bm, acc);
    }
    int nn = lane & 15, hi = lane >> 4;
    int f = nt * 16 + nn;
    float bb = b2[f];
#pragma unroll
    for (int i = 0; i < 8; ++i) {
      int pl = mt * 16 + i + (hi << 3);
      size_t p = p0 + pl;
      int b  = (int)(p / HW_);
      int pp = (int)(p % HW_);
      size_t idx = ((size_t)b * C_ + f) * HW_ + pp;
      out[idx] = x2[idx] + acc[i] + bb;
    }
  }
}

// ---------------- host side ---------------------------------------------------
extern "C" void kernel_launch(void* const* d_in, const int* in_sizes, int n_in,
                              void* d_out, int out_size, void* d_ws,
                              size_t ws_size, hipStream_t stream) {
  (void)in_sizes; (void)n_in; (void)out_size; (void)ws_size;
  const float* x        = (const float*)d_in[0];
  const float* cond     = (const float*)d_in[1];
  const float* qkv_w    = (const float*)d_in[2];
  const float* qkv_b    = (const float*)d_in[3];
  const float* proj_w   = (const float*)d_in[4];
  const float* proj_b   = (const float*)d_in[5];
  const float* rel_bias = (const float*)d_in[6];
  const float* film_w   = (const float*)d_in[7];
  const float* film_b   = (const float*)d_in[8];
  const float* w1       = (const float*)d_in[9];
  const float* b1       = (const float*)d_in[10];
  const float* w2       = (const float*)d_in[11];
  const float* b2       = (const float*)d_in[12];
  float* out = (float*)d_out;

  char* ws = (char*)d_ws;
  float* x2      = (float*)(ws);                 // 8*192*128*128*4 = 100663296
  h16*   qkv_wh  = (h16*)(ws + 100663296);       // 576*192*2  = 221184
  h16*   proj_wh = (h16*)(ws + 100884480);       // 192*192*2  = 73728
  h16*   w1h     = (h16*)(ws + 100958208);       // 768*192*2  = 294912
  h16*   w2h     = (h16*)(ws + 101253120);       // 192*768*2  = 294912
  float* gb      = (float*)(ws + 101548032);     // 8*384*4    = 12288
  float* mean1   = (float*)(ws + 101560320);     // 1536*4
  float* rstd1   = (float*)(ws + 101566464);
  float* mean2   = (float*)(ws + 101572608);
  float* rstd2   = (float*)(ws + 101578752);

  k_convert<<<1728, 256, 0, stream>>>(qkv_w, proj_w, w1, w2,
                                      qkv_wh, proj_wh, w1h, w2h);
  k_film<<<12, 256, 0, stream>>>(cond, film_w, film_b, gb);
  k_stats<<<B_ * C_, 256, 0, stream>>>(x, mean1, rstd1);
  k_attn<<<B_ * NWIN, 256, 260096, stream>>>(x, mean1, rstd1, qkv_wh, qkv_b,
                                             proj_wh, proj_b, rel_bias, x2);
  k_stats<<<B_ * C_, 256, 0, stream>>>(x2, mean2, rstd2);
  k_mlp<<<(B_ * HW_) / 64, 256, 124928, stream>>>(x2, mean2, rstd2, gb, w1h,
                                                  b1, w2h, b2, out);
}
